// GenelizedRCNN_72129680769067
// MI455X (gfx1250) — matrix operand order repaired and checked by reading, not compiled, not run on hardware
//
#include <hip/hip_runtime.h>
#include <math.h>

typedef __attribute__((ext_vector_type(16))) _Float16 v16h;
typedef __attribute__((ext_vector_type(8)))  float    v8f;
typedef __attribute__((ext_vector_type(4)))  unsigned u32x4;

#define NB    2               /* batch */
#define TOTA  279279          /* total anchors per image */
#define NEGV  (-1000000000.0f)
#define BCLIP 4.135166556742356f   /* log(1000/16) */

__constant__ int   c_LH[5]   = {208,104,52,26,13};
__constant__ int   c_LW[5]   = {336,168,84,42,21};
__constant__ int   c_LOFF[5] = {0,209664,262080,275184,278460};
__constant__ int   c_LK[5]   = {1000,1000,1000,1000,819};
__constant__ float c_LSTR[5] = {4.f,8.f,16.f,32.f,64.f};
__constant__ float c_SQA[3]  = {1.0f,0.70710678f,1.41421356f};

union Frag16 { u32x4 u[2]; v16h v; };

__device__ inline v8f zero8(){
  v8f r;
  #pragma unroll
  for (int i=0;i<8;++i) r[i]=0.f;
  return r;
}

__device__ inline unsigned fkey(float f){
  unsigned u = __float_as_uint(f);
  return (u & 0x80000000u) ? ~u : (u | 0x80000000u);
}

/* -------- block-wide argmax with first-index tie break (jnp.argmax) ------- */
__device__ int block_argmax(const float* sc, int N, float* rsv, int* riv, float* val){
  int tid = threadIdx.x, nt = blockDim.x;
  float best = -INFINITY; int bi = 0;
  for (int t = tid; t < N; t += nt){ float v = sc[t]; if (v > best){ best = v; bi = t; } }
  rsv[tid] = best; riv[tid] = bi;
  __syncthreads();
  for (int s = nt >> 1; s > 0; s >>= 1){
    if (tid < s){
      if (rsv[tid+s] > rsv[tid] || (rsv[tid+s] == rsv[tid] && riv[tid+s] < riv[tid])){
        rsv[tid] = rsv[tid+s]; riv[tid] = riv[tid+s];
      }
    }
    __syncthreads();
  }
  *val = rsv[0]; int j = riv[0];
  __syncthreads();
  return j;
}

/* ---- weight pre-swizzle into WMMA B-fragment layout: k = (e&7) + 8*hi + 16*(e>>3) */
__global__ void make_bfrag(const float* __restrict__ W, int K, int N, int ntTotal,
                           _Float16* __restrict__ out, int total){
  int t = blockIdx.x * 256 + threadIdx.x;
  if (t >= total) return;
  int e    = t & 15;
  int lane = (t >> 4) & 31;
  int rest = t >> 9;
  int nt   = rest % ntTotal;
  int ks   = rest / ntTotal;
  int hi   = lane >> 4;
  int k = ks*32 + (e & 7) + 8*hi + 16*(e >> 3);
  int n = nt*16 + (lane & 15);
  float v = (k < K && n < N) ? W[(size_t)k * N + n] : 0.f;
  out[t] = (_Float16)v;
}

/* ---------------- fragment load / issue helpers --------------------------- */
__device__ inline void conv_loadA(const float* __restrict__ feat, int b, int H, int W,
                                  int y, int xin, int ks, int hi, Frag16& a){
  int tap = ks >> 3, cstep = ks & 7;
  int ky = tap / 3, kx = tap % 3;
  int yy = y + ky - 1;
  int xx = xin + kx - 1;
  if (yy >= 0 && yy < H && xx >= 0 && xx < W){
    const float* src = feat + (((size_t)b*H + yy)*W + xx)*256 + cstep*32;
    #pragma unroll
    for (int e = 0; e < 8; ++e){
      a.v[e]   = (_Float16)src[8*hi + e];        /* K = 8*hi + e      */
      a.v[e+8] = (_Float16)src[16 + 8*hi + e];   /* K = 16 + 8*hi + e */
    }
  } else {
    #pragma unroll
    for (int e = 0; e < 16; ++e) a.v[e] = (_Float16)0.f;
  }
}

__device__ inline void conv_loadB(const _Float16* __restrict__ wfrag, int ks,
                                  int wave, int lane, Frag16 bf[4]){
  #pragma unroll
  for (int nt = 0; nt < 4; ++nt){
    const _Float16* bp = wfrag + (((size_t)ks*16 + wave*4 + nt)*32 + lane)*16;
    bf[nt].u[0] = *(const u32x4*)bp;
    bf[nt].u[1] = *(const u32x4*)(bp + 8);
  }
}

__device__ inline void conv_mma(v8f acc[4], const Frag16& a, const Frag16 bf[4]){
  #pragma unroll
  for (int nt = 0; nt < 4; ++nt)
    acc[nt] = __builtin_amdgcn_wmma_f32_16x16x32_f16(false, a.v, false, bf[nt].v,
                                                     (short)0, acc[nt], false, false);
}

__device__ inline void gemm_loadA(const _Float16* __restrict__ arow, int ks, int hi, Frag16& a){
  const _Float16* ap = arow + ks*32 + 8*hi;
  a.u[0] = *(const u32x4*)ap;          /* K = 8*hi + 0..7       */
  a.u[1] = *(const u32x4*)(ap + 16);   /* K = 16 + 8*hi + 0..7  */
}

__device__ inline void gemm_loadB(const _Float16* __restrict__ Bf, int ks, int ntTotal,
                                  int ntb, int lane, Frag16 bf[4]){
  #pragma unroll
  for (int t = 0; t < 4; ++t){
    int nt = ntb + t;
    if (nt >= ntTotal) break;          /* wave-uniform */
    const _Float16* bp = Bf + (((size_t)ks*ntTotal + nt)*32 + lane)*16;
    bf[t].u[0] = *(const u32x4*)bp;
    bf[t].u[1] = *(const u32x4*)(bp + 8);
  }
}

__device__ inline void gemm_mma(v8f acc[4], const Frag16& a, const Frag16 bf[4],
                                int ntTotal, int ntb){
  #pragma unroll
  for (int t = 0; t < 4; ++t){
    if (ntb + t >= ntTotal) break;     /* wave-uniform */
    acc[t] = __builtin_amdgcn_wmma_f32_16x16x32_f16(false, a.v, false, bf[t].v,
                                                    (short)0, acc[t], false, false);
  }
}

/* -------- fused RPN 3x3 conv (WMMA) + ReLU + score/box heads --------------
 * K-loop is 72 steps (9 taps x 8 channel blocks), software pipelined with
 * ping-pong register fragments so fragment loads for step k+1 overlap the
 * four WMMAs of step k.                                                     */
__global__ __launch_bounds__(128)
void rpn_conv_head(const float* __restrict__ feat, const _Float16* __restrict__ wfrag,
                   const float* __restrict__ cb,
                   const float* __restrict__ sw, const float* __restrict__ sb,
                   const float* __restrict__ bw, const float* __restrict__ bb,
                   float* __restrict__ scoreOut, float* __restrict__ deltaOut,
                   int H, int W, int baseOff){
  int tid  = threadIdx.x;
  int wave = tid >> 5, lane = tid & 31, hi = lane >> 4, lr = lane & 15;
  int x0 = blockIdx.x * 16, y = blockIdx.y, b = blockIdx.z;

  __shared__ _Float16 hs[16][256];
  __shared__ float s_sw[256*3];
  __shared__ float s_bw[256*12];
  for (int t = tid; t < 768;  t += 128) s_sw[t] = sw[t];
  for (int t = tid; t < 3072; t += 128) s_bw[t] = bw[t];

  v8f acc[4];
  #pragma unroll
  for (int t = 0; t < 4; ++t) acc[t] = zero8();

  int xin = x0 + lr;   /* this lane's A-row = output x position */

  Frag16 a0, a1, b0[4], b1[4];
  conv_loadA(feat, b, H, W, y, xin, 0, hi, a0);
  conv_loadB(wfrag, 0, wave, lane, b0);
  for (int ks = 0; ks < 72; ks += 2){
    int k1 = ks + 1;                       /* always < 72 (72 even) */
    conv_loadA(feat, b, H, W, y, xin, k1, hi, a1);
    conv_loadB(wfrag, k1, wave, lane, b1);
    conv_mma(acc, a0, b0);
    int k2 = (ks + 2 < 72) ? ks + 2 : ks;  /* clamp tail prefetch */
    conv_loadA(feat, b, H, W, y, xin, k2, hi, a0);
    conv_loadB(wfrag, k2, wave, lane, b0);
    conv_mma(acc, a1, b1);
  }

  /* bias + relu -> LDS activation tile */
  #pragma unroll
  for (int nt = 0; nt < 4; ++nt){
    int ntile = wave*4 + nt;
    #pragma unroll
    for (int v = 0; v < 8; ++v){
      int r  = v + 8*hi;
      int co = ntile*16 + lr;
      float val = acc[nt][v] + cb[co];
      hs[r][co] = (_Float16)fmaxf(val, 0.f);
    }
  }
  __syncthreads();

  /* 15 head outputs per pixel (3 scores + 12 deltas) */
  for (int t = tid; t < 240; t += 128){
    int m = t / 15, o = t % 15;
    int x = x0 + m;
    if (x >= W) continue;
    float accv;
    if (o < 3){
      accv = sb[o];
      for (int c = 0; c < 256; ++c) accv += (float)hs[m][c] * s_sw[c*3 + o];
    } else {
      int j = o - 3;
      accv = bb[j];
      for (int c = 0; c < 256; ++c) accv += (float)hs[m][c] * s_bw[c*12 + j];
    }
    int pos = y*W + x;
    if (o < 3)
      scoreOut[(size_t)b*TOTA + baseOff + pos*3 + o] = 1.f/(1.f + expf(-accv));
    else
      deltaOut[((size_t)b*TOTA + baseOff + pos*3)*4 + (o-3)] = accv;
  }
}

/* -------- per (image,level) radix-threshold top-k select ------------------ */
__global__ void topk_select(const float* __restrict__ scoreBuf, int* __restrict__ selIdx){
  int l = blockIdx.x, b = blockIdx.y;
  int N = c_LH[l]*c_LW[l]*3;
  int k = c_LK[l];
  const float* s = scoreBuf + (size_t)b*TOTA + c_LOFF[l];
  int* out = selIdx + (b*5 + l)*1000;
  __shared__ int hist[4096];
  __shared__ int sv[8];
  int tid = threadIdx.x, nt = blockDim.x;

  for (int i = tid; i < 4096; i += nt) hist[i] = 0;
  __syncthreads();
  for (int t = tid; t < N; t += nt){ unsigned u = fkey(s[t]); atomicAdd(&hist[u >> 20], 1); }
  __syncthreads();
  if (tid == 0){
    int cum = 0;
    for (int i = 4095; i >= 0; --i){
      cum += hist[i];
      if (cum >= k){ sv[0] = i; sv[1] = k - (cum - hist[i]); sv[2] = cum - hist[i]; break; }
    }
  }
  __syncthreads();
  int t1 = sv[0], kRem = sv[1], before1 = sv[2];

  for (int i = tid; i < 4096; i += nt) hist[i] = 0;
  __syncthreads();
  for (int t = tid; t < N; t += nt){
    unsigned u = fkey(s[t]);
    if ((int)(u >> 20) == t1) atomicAdd(&hist[(u >> 8) & 0xFFF], 1);
  }
  __syncthreads();
  if (tid == 0){
    int cum = 0;
    for (int i = 4095; i >= 0; --i){
      cum += hist[i];
      if (cum >= kRem){ sv[3] = i; sv[4] = kRem - (cum - hist[i]); sv[5] = before1 + (cum - hist[i]); break; }
    }
    sv[6] = 0; sv[7] = 0;
  }
  __syncthreads();
  int t2 = sv[3], kRem2 = sv[4], cntAbove = sv[5];

  for (int t = tid; t < N; t += nt){
    unsigned u = fkey(s[t]);
    int b1 = u >> 20;
    if (b1 > t1){ int p = atomicAdd(&sv[6], 1); out[p] = t; }
    else if (b1 == t1){
      int b2 = (u >> 8) & 0xFFF;
      if (b2 > t2){ int p = atomicAdd(&sv[6], 1); out[p] = t; }
      else if (b2 == t2){ int p = atomicAdd(&sv[7], 1); if (p < kRem2) out[cntAbove + p] = t; }
    }
  }
}

/* -------- decode RPN anchors for selected proposals ----------------------- */
__global__ void rpn_decode(const int* __restrict__ selIdx, const float* __restrict__ scoreBuf,
                           const float* __restrict__ deltaBuf, const float* __restrict__ imInfo,
                           float* __restrict__ decB, float* __restrict__ decS){
  int t = blockIdx.x*blockDim.x + threadIdx.x;
  if (t >= NB*5*1000) return;
  int slot = t % 1000, l = (t/1000) % 5, b = t/5000;
  if (slot >= c_LK[l]){
    decS[t] = NEGV;
    decB[t*4+0]=0.f; decB[t*4+1]=0.f; decB[t*4+2]=0.f; decB[t*4+3]=0.f;
    return;
  }
  int ai = selIdx[t];
  size_t gi = (size_t)b*TOTA + c_LOFF[l] + ai;
  float sc = scoreBuf[gi];
  float d0 = deltaBuf[gi*4+0], d1 = deltaBuf[gi*4+1];
  float d2 = deltaBuf[gi*4+2], d3 = deltaBuf[gi*4+3];
  int W = c_LW[l];
  int pos = ai/3, a = ai%3;
  int y = pos / W, x = pos % W;
  float stride = c_LSTR[l], size = 8.f*stride;
  float cy = (y + 0.5f)*stride, cx = (x + 0.5f)*stride;
  float sr = c_SQA[a];
  float ah = size/sr, aw = size*sr;
  float dh = fminf(d2, BCLIP), dw = fminf(d3, BCLIP);
  float pcy = d0*ah + cy, pcx = d1*aw + cx;
  float ph = expf(dh)*ah, pw = expf(dw)*aw;
  float hI = imInfo[b*5+0], wI = imInfo[b*5+1];
  decB[t*4+0] = fminf(fmaxf(pcy - 0.5f*ph, 0.f), hI);
  decB[t*4+1] = fminf(fmaxf(pcx - 0.5f*pw, 0.f), wI);
  decB[t*4+2] = fminf(fmaxf(pcy + 0.5f*ph, 0.f), hI);
  decB[t*4+3] = fminf(fmaxf(pcx + 0.5f*pw, 0.f), wI);
  decS[t] = sc;
}

/* -------- generic block NMS (matches nms_single semantics) ---------------- */
__global__ __launch_bounds__(256)
void nms_kernel(const float* __restrict__ boxesAll, const float* __restrict__ scoresAll,
                int N, float thresh, int maxOut,
                float* __restrict__ outB, float* __restrict__ outS){
  int list = blockIdx.x, tid = threadIdx.x;
  const float* bsrc = boxesAll + (size_t)list*N*4;
  const float* ssrc = scoresAll + (size_t)list*N;
  float* ob = outB + (size_t)list*maxOut*4;
  float* os = outS + (size_t)list*maxOut;
  __shared__ float sc[1024];
  __shared__ float bx[1024][4];
  __shared__ float rsv[256];
  __shared__ int   riv[256];
  for (int t = tid; t < N; t += 256){
    sc[t] = ssrc[t];
    bx[t][0]=bsrc[t*4+0]; bx[t][1]=bsrc[t*4+1]; bx[t][2]=bsrc[t*4+2]; bx[t][3]=bsrc[t*4+3];
  }
  __syncthreads();
  for (int i = 0; i < maxOut; ++i){
    float bs; int j = block_argmax(sc, N, rsv, riv, &bs);
    if (bs <= NEGV*0.5f){
      for (int t = i + tid; t < maxOut; t += 256){
        os[t] = NEGV;
        ob[t*4+0]=0.f; ob[t*4+1]=0.f; ob[t*4+2]=0.f; ob[t*4+3]=0.f;
      }
      break;
    }
    float y1 = bx[j][0], x1 = bx[j][1], y2 = bx[j][2], x2 = bx[j][3];
    if (tid == 0){
      os[i] = bs;
      ob[i*4+0]=y1; ob[i*4+1]=x1; ob[i*4+2]=y2; ob[i*4+3]=x2;
    }
    float area = (y2 - y1)*(x2 - x1);
    for (int t = tid; t < N; t += 256){
      float yy1 = fmaxf(y1, bx[t][0]), xx1 = fmaxf(x1, bx[t][1]);
      float yy2 = fminf(y2, bx[t][2]), xx2 = fminf(x2, bx[t][3]);
      float inter = fmaxf(yy2 - yy1, 0.f)*fmaxf(xx2 - xx1, 0.f);
      float ab = (bx[t][2]-bx[t][0])*(bx[t][3]-bx[t][1]);
      float iou = inter / (area + ab - inter + 1e-8f);
      if (iou >= thresh) sc[t] = NEGV;
    }
    __syncthreads();
    if (tid == 0) sc[j] = NEGV;
    __syncthreads();
  }
}

/* -------- merge 5 level NMS lists -> top-1000 rois ------------------------ */
__global__ __launch_bounds__(256)
void merge_rois(const float* __restrict__ nmsS, const float* __restrict__ nmsB,
                float* __restrict__ rois){
  int b = blockIdx.x, tid = threadIdx.x;
  __shared__ float sc[5120];
  __shared__ float rsv[256];
  __shared__ int   riv[256];
  for (int t = tid; t < 5120; t += 256)
    sc[t] = (t < 5000) ? nmsS[(size_t)b*5000 + t] : -INFINITY;
  __syncthreads();
  for (int i = 0; i < 1000; ++i){
    float v; int j = block_argmax(sc, 5000, rsv, riv, &v);
    if (tid == 0){
      for (int c = 0; c < 4; ++c)
        rois[((size_t)b*1000 + i)*4 + c] = nmsB[((size_t)b*5000 + j)*4 + c];
      sc[j] = -INFINITY;
    }
    __syncthreads();
  }
}

/* -------- multilevel ROI align (writes f16 im2row for the MLP) ------------ */
__global__ __launch_bounds__(256)
void roi_align(const float* __restrict__ rois,
               const float* __restrict__ f0, const float* __restrict__ f1,
               const float* __restrict__ f2, const float* __restrict__ f3,
               const float* __restrict__ f4,
               _Float16* __restrict__ roiF){
  int cell = blockIdx.x, k = blockIdx.y, b = blockIdx.z, c = threadIdx.x;
  int oy = cell / 7, ox = cell % 7;
  const float* r = rois + ((size_t)b*1000 + k)*4;
  float y1 = r[0], x1 = r[1], y2 = r[2], x2 = r[3];
  float area = fmaxf((y2 - y1)*(x2 - x1), 1e-6f);
  float lvlf = floorf(4.f + log2f(sqrtf(area)/224.f));
  lvlf = fminf(fmaxf(lvlf, 2.f), 6.f);
  int li = (int)lvlf - 2;
  int H = c_LH[li], W = c_LW[li];
  float stride = c_LSTR[li];
  const float* f = (li==0)?f0:(li==1)?f1:(li==2)?f2:(li==3)?f3:f4;
  float uy = (oy + 0.5f)/7.f, ux = (ox + 0.5f)/7.f;
  float ys = (y1 + (y2 - y1)*uy)/stride - 0.5f;
  float xs = (x1 + (x2 - x1)*ux)/stride - 0.5f;
  ys = fminf(fmaxf(ys, 0.f), (float)H - 1.f);
  xs = fminf(fmaxf(xs, 0.f), (float)W - 1.f);
  float y0f = floorf(ys), x0f = floorf(xs);
  float wy = ys - y0f, wx = xs - x0f;
  int y0i = (int)y0f, x0i = (int)x0f;
  int y1i = min(y0i + 1, H - 1), x1i = min(x0i + 1, W - 1);
  size_t base = (size_t)b*H*W*256;
  float g00 = f[base + ((size_t)y0i*W + x0i)*256 + c];
  float g01 = f[base + ((size_t)y0i*W + x1i)*256 + c];
  float g10 = f[base + ((size_t)y1i*W + x0i)*256 + c];
  float g11 = f[base + ((size_t)y1i*W + x1i)*256 + c];
  float o = (1.f-wy)*(1.f-wx)*g00 + (1.f-wy)*wx*g01 + wy*(1.f-wx)*g10 + wy*wx*g11;
  roiF[(((size_t)b*1000 + k)*49 + cell)*256 + c] = (_Float16)o;
}

/* -------- generic f16-in / f32-acc WMMA GEMM (pipelined) ------------------
 * One wave computes a 16x64 tile (4 N-tiles). KS is even for all call sites,
 * so the K loop is unrolled by two with ping-pong fragment buffers.          */
__global__ __launch_bounds__(128)
void wmma_gemm(const _Float16* __restrict__ A, int lda,
               const _Float16* __restrict__ Bf, int ntTotal,
               const float* __restrict__ bias,
               float* __restrict__ outF, _Float16* __restrict__ outH,
               int ldc, int Nreal, int K, int relu){
  int tid = threadIdx.x, wave = tid >> 5, lane = tid & 31, hi = lane >> 4, lr = lane & 15;
  int m0  = blockIdx.y * 16;
  int ntb = (blockIdx.x*4 + wave)*4;
  v8f acc[4];
  #pragma unroll
  for (int t = 0; t < 4; ++t) acc[t] = zero8();

  const _Float16* arow = A + (size_t)(m0 + lr)*lda;
  const int KS = K >> 5;

  Frag16 a0, a1, b0[4], b1[4];
  gemm_loadA(arow, 0, hi, a0);
  gemm_loadB(Bf, 0, ntTotal, ntb, lane, b0);
  for (int ks = 0; ks < KS; ks += 2){
    int k1 = ks + 1;                       /* always < KS (KS even) */
    gemm_loadA(arow, k1, hi, a1);
    gemm_loadB(Bf, k1, ntTotal, ntb, lane, b1);
    gemm_mma(acc, a0, b0, ntTotal, ntb);
    int k2 = (ks + 2 < KS) ? ks + 2 : ks;  /* clamp tail prefetch */
    gemm_loadA(arow, k2, hi, a0);
    gemm_loadB(Bf, k2, ntTotal, ntb, lane, b0);
    gemm_mma(acc, a1, b1, ntTotal, ntb);
  }

  #pragma unroll
  for (int t = 0; t < 4; ++t){
    int nt = ntb + t;
    if (nt >= ntTotal) break;
    #pragma unroll
    for (int v = 0; v < 8; ++v){
      int r = m0 + v + 8*hi;
      int col = nt*16 + lr;
      if (col >= Nreal) continue;
      float val = acc[t][v] + (bias ? bias[col] : 0.f);
      if (relu) val = fmaxf(val, 0.f);
      if (outH) outH[(size_t)r*ldc + col] = (_Float16)val;
      else      outF[(size_t)r*ldc + col] = val;
    }
  }
}

/* -------- per-roi softmax + per-class box decode -------------------------- */
__global__ void det_decode(const float* __restrict__ clsLog, const float* __restrict__ boxOut,
                           const float* __restrict__ rois, const float* __restrict__ imInfo,
                           float* __restrict__ fgS, float* __restrict__ clsBox){
  int t = blockIdx.x*blockDim.x + threadIdx.x;
  if (t >= NB*1000) return;
  int b = t / 1000, k = t % 1000;
  const float* lg = clsLog + (size_t)t*96;
  float m = lg[0];
  for (int c = 1; c < 91; ++c) m = fmaxf(m, lg[c]);
  float denom = 0.f;
  for (int c = 0; c < 91; ++c) denom += expf(lg[c] - m);
  float inv = 1.f / denom;

  const float* r = rois + (size_t)t*4;
  float y1 = r[0], x1 = r[1], y2 = r[2], x2 = r[3];
  float ha = y2 - y1, wa = x2 - x1;
  float yc = y1 + 0.5f*ha, xc = x1 + 0.5f*wa;
  float hI = imInfo[b*5+0], wI = imInfo[b*5+1];
  const float* d = boxOut + (size_t)t*368;

  for (int c = 1; c < 91; ++c){
    float p = expf(lg[c] - m)*inv;
    float dy = d[c*4+0]/10.f, dx = d[c*4+1]/10.f;
    float dh = fminf(d[c*4+2]/5.f, BCLIP), dw = fminf(d[c*4+3]/5.f, BCLIP);
    float pcy = dy*ha + yc, pcx = dx*wa + xc;
    float ph = expf(dh)*ha, pw = expf(dw)*wa;
    size_t o = ((size_t)b*90 + (c-1))*1000 + k;
    fgS[o] = (p >= 0.05f) ? p : NEGV;
    clsBox[o*4+0] = fminf(fmaxf(pcy - 0.5f*ph, 0.f), hI);
    clsBox[o*4+1] = fminf(fmaxf(pcx - 0.5f*pw, 0.f), wI);
    clsBox[o*4+2] = fminf(fmaxf(pcy + 0.5f*ph, 0.f), hI);
    clsBox[o*4+3] = fminf(fmaxf(pcx + 0.5f*pw, 0.f), wI);
  }
}

/* -------- final top-100 over 90 class NMS lists --------------------------- */
__global__ __launch_bounds__(256)
void final_dets(const float* __restrict__ nmsS, const float* __restrict__ nmsB,
                float* __restrict__ out){
  int b = blockIdx.x, tid = threadIdx.x;
  __shared__ float sc[9216];
  __shared__ float rsv[256];
  __shared__ int   riv[256];
  for (int t = tid; t < 9216; t += 256)
    sc[t] = (t < 9000) ? nmsS[(size_t)b*9000 + t] : -INFINITY;
  __syncthreads();
  int cnt = 0;
  float* ob  = out + NB + (size_t)b*400;
  float* oc  = out + NB + NB*400 + (size_t)b*100;
  float* osv = out + NB + NB*500 + (size_t)b*100;
  for (int i = 0; i < 100; ++i){
    float val; int j = block_argmax(sc, 9000, rsv, riv, &val);
    bool valid = val > 0.f;
    if (valid) cnt++;
    if (tid == 0){
      osv[i] = valid ? val : 0.f;
      oc[i]  = valid ? (float)(j/100 + 1) : 0.f;
      for (int c = 0; c < 4; ++c)
        ob[i*4 + c] = valid ? nmsB[((size_t)b*9000 + j)*4 + c] : 0.f;
      sc[j] = -INFINITY;
    }
    __syncthreads();
  }
  if (tid == 0) out[b] = (float)cnt;
}

/* ========================== host driver =================================== */
extern "C" void kernel_launch(void* const* d_in, const int* in_sizes, int n_in,
                              void* d_out, int out_size, void* d_ws, size_t ws_size,
                              hipStream_t stream){
  (void)in_sizes; (void)n_in; (void)out_size; (void)ws_size;
  const float* feats[5] = {(const float*)d_in[0], (const float*)d_in[1], (const float*)d_in[2],
                           (const float*)d_in[3], (const float*)d_in[4]};
  const float* image_info  = (const float*)d_in[5];
  const float* rpn_conv_w  = (const float*)d_in[6];
  const float* rpn_conv_b  = (const float*)d_in[7];
  const float* rpn_score_w = (const float*)d_in[8];
  const float* rpn_score_b = (const float*)d_in[9];
  const float* rpn_box_w   = (const float*)d_in[10];
  const float* rpn_box_b   = (const float*)d_in[11];
  const float* fc1_w = (const float*)d_in[12];
  const float* fc1_b = (const float*)d_in[13];
  const float* fc2_w = (const float*)d_in[14];
  const float* fc2_b = (const float*)d_in[15];
  const float* cls_w = (const float*)d_in[16];
  const float* cls_b = (const float*)d_in[17];
  const float* box_w = (const float*)d_in[18];
  const float* box_b = (const float*)d_in[19];
  float* out = (float*)d_out;

  char* base = (char*)d_ws; size_t off = 0;
  auto alloc = [&](size_t bytes)->void*{
    size_t a = (off + 255) & ~(size_t)255;
    off = a + bytes;
    return (void*)(base + a);
  };
  _Float16* convF = (_Float16*)alloc((size_t)589824*2);
  _Float16* fc1F  = (_Float16*)alloc((size_t)12845056*2);
  _Float16* fc2F  = (_Float16*)alloc((size_t)1048576*2);
  _Float16* clsF  = (_Float16*)alloc((size_t)98304*2);
  _Float16* boxF  = (_Float16*)alloc((size_t)376832*2);
  float* scoreBuf = (float*)alloc((size_t)NB*TOTA*4);
  float* deltaBuf = (float*)alloc((size_t)NB*TOTA*4*4);
  int*   selIdx   = (int*)alloc((size_t)NB*5*1000*4);
  float* decB     = (float*)alloc((size_t)NB*5*1000*4*4);
  float* decS     = (float*)alloc((size_t)NB*5*1000*4);
  float* nmsB     = (float*)alloc((size_t)NB*5*1000*4*4);
  float* nmsS     = (float*)alloc((size_t)NB*5*1000*4);
  float* rois     = (float*)alloc((size_t)NB*1000*4*4);
  _Float16* roiF  = (_Float16*)alloc((size_t)2000*12544*2);
  _Float16* h1    = (_Float16*)alloc((size_t)2000*1024*2);
  _Float16* h2    = (_Float16*)alloc((size_t)2000*1024*2);
  float* clsLog   = (float*)alloc((size_t)2000*96*4);
  float* boxOut   = (float*)alloc((size_t)2000*368*4);
  float* fgS      = (float*)alloc((size_t)NB*90*1000*4);
  float* clsBox   = (float*)alloc((size_t)NB*90*1000*4*4);
  float* dNmsB    = (float*)alloc((size_t)NB*90*100*4*4);
  float* dNmsS    = (float*)alloc((size_t)NB*90*100*4);

  auto cdiv = [](int a, int b){ return (a + b - 1)/b; };

  /* weight pre-swizzle into WMMA B-fragment layouts */
  make_bfrag<<<cdiv(589824,256),256,0,stream>>>(rpn_conv_w, 2304,  256, 16, convF, 589824);
  make_bfrag<<<cdiv(12845056,256),256,0,stream>>>(fc1_w,   12544, 1024, 64, fc1F, 12845056);
  make_bfrag<<<cdiv(1048576,256),256,0,stream>>>(fc2_w,     1024, 1024, 64, fc2F, 1048576);
  make_bfrag<<<cdiv(98304,256),256,0,stream>>>(cls_w,       1024,   91,  6, clsF, 98304);
  make_bfrag<<<cdiv(376832,256),256,0,stream>>>(box_w,      1024,  364, 23, boxF, 376832);

  int LH[5] = {208,104,52,26,13}, LW[5] = {336,168,84,42,21};
  int LOFFh[5] = {0,209664,262080,275184,278460};
  for (int l = 0; l < 5; ++l){
    dim3 g(cdiv(LW[l],16), LH[l], NB);
    rpn_conv_head<<<g,128,0,stream>>>(feats[l], convF, rpn_conv_b,
                                      rpn_score_w, rpn_score_b, rpn_box_w, rpn_box_b,
                                      scoreBuf, deltaBuf, LH[l], LW[l], LOFFh[l]);
  }
  topk_select<<<dim3(5,NB),512,0,stream>>>(scoreBuf, selIdx);
  rpn_decode<<<cdiv(NB*5000,256),256,0,stream>>>(selIdx, scoreBuf, deltaBuf, image_info, decB, decS);
  nms_kernel<<<NB*5,256,0,stream>>>(decB, decS, 1000, 0.7f, 1000, nmsB, nmsS);
  merge_rois<<<NB,256,0,stream>>>(nmsS, nmsB, rois);
  roi_align<<<dim3(49,1000,NB),256,0,stream>>>(rois, feats[0], feats[1], feats[2], feats[3], feats[4], roiF);

  wmma_gemm<<<dim3(4,125),128,0,stream>>>(roiF, 12544, fc1F, 64, fc1_b, (float*)nullptr, h1, 1024, 1024, 12544, 1);
  wmma_gemm<<<dim3(4,125),128,0,stream>>>(h1,   1024,  fc2F, 64, fc2_b, (float*)nullptr, h2, 1024, 1024, 1024, 1);
  wmma_gemm<<<dim3(1,125),128,0,stream>>>(h2,   1024,  clsF,  6, cls_b, clsLog, (_Float16*)nullptr, 96, 91, 1024, 0);
  wmma_gemm<<<dim3(2,125),128,0,stream>>>(h2,   1024,  boxF, 23, box_b, boxOut, (_Float16*)nullptr, 368, 364, 1024, 0);

  det_decode<<<cdiv(NB*1000,256),256,0,stream>>>(clsLog, boxOut, rois, image_info, fgS, clsBox);
  nms_kernel<<<NB*90,256,0,stream>>>(clsBox, fgS, 1000, 0.5f, 100, dNmsB, dNmsS);
  final_dets<<<NB,256,0,stream>>>(dNmsS, dNmsB, out);
}
